// NeuralSDEHead_14774687498271
// MI455X (gfx1250) — matrix operand order repaired and checked by reading, not compiled, not run on hardware
//
#include <hip/hip_runtime.h>
#include <hip/hip_bf16.h>

typedef _Float16 half16 __attribute__((ext_vector_type(16)));
typedef float    float8 __attribute__((ext_vector_type(8)));

#define BATCH 8
#define DMODEL 512
#define HID 64
#define HOR 128
#define NP 4096
#define PPW 16                 // paths per wave
#define NWAVES ((BATCH*NP)/PPW)  // 2048
#define DT 1.0f
#define SQRT_DT 1.0f
#define PATHS_ELEMS (BATCH*NP*HOR)

// Hardware-rate transcendentals: v_rcp_f32 / v_exp_f32 / v_log_f32 / v_sqrt_f32 / v_tanh_f32.
__device__ __forceinline__ float fast_rcp(float x)  { return __builtin_amdgcn_rcpf(x); }
__device__ __forceinline__ float fast_sqrt(float x) { return __builtin_amdgcn_sqrtf(x); }

// silu(x) = x*sigmoid(x) = 0.5x*tanh(0.5x) + 0.5x (exact identity).
// gfx1250 has a hardware TANH trans op -> 1 trans + 2 VALU instead of
// exp + add + rcp + mul (2 trans + 2 VALU + hazard bubbles).
#if __has_builtin(__builtin_amdgcn_tanhf)
__device__ __forceinline__ float silu_f(float x) {
    float h = 0.5f * x;
    return __fmaf_rn(h, __builtin_amdgcn_tanhf(h), h);
}
#else
__device__ __forceinline__ float silu_f(float x) {
    return x * fast_rcp(1.0f + __expf(-x));
}
#endif
__device__ __forceinline__ float softplus_f(float x) {
    return (x > 20.0f) ? x : __logf(1.0f + __expf(x));
}
// Counter-based Gaussian (splitmix-style hash + Box-Muller). Deterministic in (gid, t).
__device__ __forceinline__ float rnormal(unsigned gid, int t) {
    unsigned long long s = ((unsigned long long)gid << 32)
                         ^ (unsigned long long)((unsigned)t * 0x9E3779B9u) ^ 42ull;
    s *= 0x9E3779B97F4A7C15ull; s ^= s >> 30;
    s *= 0xBF58476D1CE4E5B9ull; s ^= s >> 27;
    s *= 0x94D049BB133111EBull; s ^= s >> 31;
    unsigned u1 = (unsigned)s, u2 = (unsigned)(s >> 32);
    float f1 = ((float)(u1 >> 8) + 0.5f) * (1.0f / 16777216.0f);
    float f2 = ((float)(u2 >> 8) + 0.5f) * (1.0f / 16777216.0f);
    float r = fast_sqrt(-2.0f * __logf(f1));
    return r * __cosf(6.28318530718f * f2);
}

// ---- Kernel 1: per-batch context projections (folds the 512-wide GEMMs once) ----
// ws[0      .. 511]  : ctxf[b][j] = h_t[b] . W_f1[:512, j] + b_f1[j]
// ws[512    .. 1023] : ctxg[b][j] = h_t[b] . W_g1[:512, j] + b_g1[j]
// out_tail[0..7] = mu, out_tail[8..15] = sigma
__global__ __launch_bounds__(64)
void sde_ctx_kernel(const float* __restrict__ h_t,
                    const float* __restrict__ W_f1, const float* __restrict__ b_f1,
                    const float* __restrict__ W_g1, const float* __restrict__ b_g1,
                    const float* __restrict__ W_mu, const float* __restrict__ b_mu,
                    const float* __restrict__ W_sig, const float* __restrict__ b_sig,
                    float* __restrict__ ws, float* __restrict__ out_tail) {
    const int b = blockIdx.x;
    const int j = threadIdx.x;
    const float* h = h_t + b * DMODEL;
    float sf = 0.f, sg = 0.f;
    for (int k = 0; k < DMODEL; ++k) {
        float hv = h[k];
        sf += hv * W_f1[k * HID + j];
        sg += hv * W_g1[k * HID + j];
    }
    ws[b * HID + j]               = sf + b_f1[j];
    ws[BATCH * HID + b * HID + j] = sg + b_g1[j];
    if (j == 0) {
        float s = 0.f;
        for (int k = 0; k < DMODEL; ++k) s += h[k] * W_mu[k];
        out_tail[b] = s + b_mu[0];
    }
    if (j == 1) {
        float s = 0.f;
        for (int k = 0; k < DMODEL; ++k) s += h[k] * W_sig[k];
        out_tail[BATCH + b] = softplus_f(s + b_sig[0]) + 1e-6f;
    }
}

// ---- Kernel 2: main SDE rollout. 1 wave per workgroup, 16 paths per wave. ----
__global__ __launch_bounds__(32)
void sde_path_kernel(const float* __restrict__ initial_price,
                     const float* __restrict__ W_f1,
                     const float* __restrict__ W_f2, const float* __restrict__ b_f2,
                     const float* __restrict__ W_f3, const float* __restrict__ b_f3,
                     const float* __restrict__ W_g1,
                     const float* __restrict__ W_g2, const float* __restrict__ b_g2,
                     const float* __restrict__ ws,
                     float* __restrict__ out) {
    __shared__ float ctxf_s[HID], ctxg_s[HID];
    __shared__ float wf1y_s[HID], wf1t_s[HID], wg1y_s[HID], wg1t_s[HID], wg2_s[HID];
    __shared__ unsigned h1u[PPW * HID / 2];   // h1 tile, f16 packed, A-operand source
    __shared__ float y_s[PPW], f_s[PPW], g_s[PPW];
    __shared__ float stage[PPW * HOR];        // raw log-paths, coalesced writeback at end

    const int lane  = threadIdx.x;
    const int w     = blockIdx.x;          // 0..2047
    const int b     = w >> 8;              // 256 waves per batch row (4096/16)
    const unsigned gidbase = (unsigned)w * PPW;  // global path id base

    // ---- one-time LDS fill of step-invariant vectors ----
    for (int j = lane; j < HID; j += 32) {
        ctxf_s[j] = ws[b * HID + j];
        ctxg_s[j] = ws[BATCH * HID + b * HID + j];
        wf1y_s[j] = W_f1[512 * HID + j];   // y-row of W_f1
        wf1t_s[j] = W_f1[513 * HID + j];   // t-row of W_f1
        wg1y_s[j] = W_g1[512 * HID + j];
        wg1t_s[j] = W_g1[513 * HID + j];
        wg2_s[j]  = W_g2[j];
    }
    if (lane < PPW) y_s[lane] = __logf(initial_price[b]);
    __syncthreads();  // single-wave WG: barrier is effectively free

    // ---- per-lane constants ----
    const int n  = lane & 15;   // C/D column, B column
    const int lg = lane >> 4;   // half-wave selector
    float wf3r[4], biasn[4];
#pragma unroll
    for (int nt = 0; nt < 4; ++nt) {
        wf3r[nt]  = W_f3[nt * 16 + n];
        biasn[nt] = b_f2[nt * 16 + n];
    }
    const float bf3 = b_f3[0], bg2 = b_g2[0];

    // ---- B fragments of W_f2 (f16), resident in VGPRs for the whole loop ----
    // B 32x16 fragment, contiguous k order matching A: lanes 0-15 hold K=0..15 (2 per VGPR), lanes 16-31 hold K=16..31.
    half16 Bf[2][4];
#pragma unroll
    for (int kt = 0; kt < 2; ++kt) {
#pragma unroll
        for (int nt = 0; nt < 4; ++nt) {
            union { unsigned u[8]; half16 v; } bu;
#pragma unroll
            for (int v = 0; v < 8; ++v) {
                int k = kt * 32 + lg * 16 + 2 * v;
                union { _Float16 h[2]; unsigned u; } pk;
                pk.h[0] = (_Float16)W_f2[k * HID + nt * 16 + n];
                pk.h[1] = (_Float16)W_f2[(k + 1) * HID + nt * 16 + n];
                bu.u[v] = pk.u;
            }
            Bf[kt][nt] = bu.v;
        }
    }

    const int p2 = lane >> 1, chunk = lane & 1, cbase = chunk * 32;

    for (int t = 0; t < HOR; ++t) {
        const float tf = (float)t * DT;
        const float y  = y_s[p2];
        float gp = 0.f;
        // h1 = silu(ctxf + y*wf1y + t*wf1t) -> LDS (f16, A layout source)
        // g partial = sum silu(ctxg + y*wg1y + t*wg1t) * wg2  (32 channels/lane)
#pragma unroll
        for (int c = 0; c < 32; c += 2) {
            int c0 = cbase + c, c1 = c0 + 1;
            float pa = ctxf_s[c0] + y * wf1y_s[c0] + tf * wf1t_s[c0];
            float pb = ctxf_s[c1] + y * wf1y_s[c1] + tf * wf1t_s[c1];
            union { _Float16 h[2]; unsigned u; } pk;
            pk.h[0] = (_Float16)silu_f(pa);
            pk.h[1] = (_Float16)silu_f(pb);
            h1u[(p2 * HID + c0) >> 1] = pk.u;
            float qa = ctxg_s[c0] + y * wg1y_s[c0] + tf * wg1t_s[c0];
            float qb = ctxg_s[c1] + y * wg1y_s[c1] + tf * wg1t_s[c1];
            gp += silu_f(qa) * wg2_s[c0] + silu_f(qb) * wg2_s[c1];
        }
        __syncthreads();

        // ---- assemble A fragments per 16-bit A 16x32 layout ----
        const int m = lane & 15;
        half16 A[2];
#pragma unroll
        for (int kt = 0; kt < 2; ++kt) {
            union { unsigned u[8]; half16 v; } au;
#pragma unroll
            for (int v = 0; v < 8; ++v) {
                int k0 = kt * 32 + lg * 16 + v * 2;
                au.u[v] = h1u[(m * HID + k0) >> 1];
            }
            A[kt] = au.v;
        }

        // ---- h2 = silu(h1 @ W_f2 + b_f2) via 8x v_wmma_f32_16x16x32_f16;
        //      immediately contract with W_f3 in the C/D layout ----
        float fp[8];
#pragma unroll
        for (int r = 0; r < 8; ++r) fp[r] = 0.f;
#pragma unroll
        for (int nt = 0; nt < 4; ++nt) {
            float bi = biasn[nt];
            float8 acc = { bi, bi, bi, bi, bi, bi, bi, bi };
            acc = __builtin_amdgcn_wmma_f32_16x16x32_f16(
                      false, A[0], false, Bf[0][nt], (short)0, acc, false, false);
            acc = __builtin_amdgcn_wmma_f32_16x16x32_f16(
                      false, A[1], false, Bf[1][nt], (short)0, acc, false, false);
#pragma unroll
            for (int r = 0; r < 8; ++r) fp[r] += silu_f(acc[r]) * wf3r[nt];
        }
        // reduce over the 16 columns (lanes of each half-wave); rows stay put
#pragma unroll
        for (int mask = 1; mask < 16; mask <<= 1) {
#pragma unroll
            for (int r = 0; r < 8; ++r) fp[r] += __shfl_xor(fp[r], mask, 16);
        }
        // g head: combine the 2 lanes per path, then softplus
        gp += __shfl_xor(gp, 1, 32);
        if (n == 0) {
#pragma unroll
            for (int r = 0; r < 8; ++r) f_s[lg * 8 + r] = fp[r] + bf3;  // row = r + 8*lg
        }
        if (chunk == 0) g_s[p2] = softplus_f(gp + bg2) + 1e-6f;
        __syncthreads();

        // ---- Euler-Maruyama update ----
        if (lane < PPW) {
            float dW = SQRT_DT * rnormal(gidbase + lane, t);
            float yn = y_s[lane] + f_s[lane] * DT + g_s[lane] * dW;
            y_s[lane] = yn;
            stage[lane * HOR + t] = yn;
        }
        __syncthreads();
    }

    // ---- coalesced writeback: clip + exp; wave's 16 paths are contiguous ----
    float* obase = out + (size_t)gidbase * HOR;
    for (int i = lane; i < PPW * HOR; i += 32) {
        float v = stage[i];
        v = fminf(20.0f, fmaxf(-20.0f, v));
        obase[i] = __expf(v);
    }
}

extern "C" void kernel_launch(void* const* d_in, const int* in_sizes, int n_in,
                              void* d_out, int out_size, void* d_ws, size_t ws_size,
                              hipStream_t stream) {
    const float* h_t   = (const float*)d_in[0];
    const float* ip    = (const float*)d_in[1];
    const float* W_f1  = (const float*)d_in[2];
    const float* b_f1  = (const float*)d_in[3];
    const float* W_f2  = (const float*)d_in[4];
    const float* b_f2  = (const float*)d_in[5];
    const float* W_f3  = (const float*)d_in[6];
    const float* b_f3  = (const float*)d_in[7];
    const float* W_g1  = (const float*)d_in[8];
    const float* b_g1  = (const float*)d_in[9];
    const float* W_g2  = (const float*)d_in[10];
    const float* b_g2  = (const float*)d_in[11];
    const float* W_mu  = (const float*)d_in[12];
    const float* b_mu  = (const float*)d_in[13];
    const float* W_sig = (const float*)d_in[14];
    const float* b_sig = (const float*)d_in[15];
    (void)in_sizes; (void)n_in; (void)out_size; (void)ws_size;

    float* out      = (float*)d_out;
    float* out_tail = out + (size_t)PATHS_ELEMS;  // mu then sigma
    float* ws       = (float*)d_ws;               // 2*8*64 floats

    sde_ctx_kernel<<<BATCH, HID, 0, stream>>>(h_t, W_f1, b_f1, W_g1, b_g1,
                                              W_mu, b_mu, W_sig, b_sig,
                                              ws, out_tail);
    sde_path_kernel<<<NWAVES, 32, 0, stream>>>(ip, W_f1, W_f2, b_f2, W_f3, b_f3,
                                               W_g1, W_g2, b_g2, ws, out);
}
